// LinearMem_16587163697535
// MI455X (gfx1250) — compile-verified, matrix operand order checked by reading, not run on hardware
//
#include <hip/hip_runtime.h>

typedef __attribute__((ext_vector_type(16))) _Float16 v16h;
typedef __attribute__((ext_vector_type(8)))  _Float16 v8h;
typedef __attribute__((ext_vector_type(8)))  float    v8f;

// ---------------------------------------------------------------------------
// Kernel 1: fake-quantize activations x [M, K] per (1,32) block -> f16
// ---------------------------------------------------------------------------
__global__ __launch_bounds__(256)
void quant_x_kernel(const float* __restrict__ x, _Float16* __restrict__ qx, int K) {
    const int row  = blockIdx.x;
    const int lane = threadIdx.x & 31;
    const int wave = threadIdx.x >> 5;
    const long base = (long)row * K;
    for (int it = 0; it < K; it += 256) {
        const int col = it + wave * 32 + lane;
        const float v = x[base + col];
        float a = fabsf(v);
        #pragma unroll
        for (int off = 16; off > 0; off >>= 1)
            a = fmaxf(a, __shfl_xor(a, off, 32));
        const float scale = (a > 0.f) ? a * (1.0f / 127.0f) : 1.0f;
        float q = rintf(v / scale);
        q = fminf(127.f, fmaxf(-127.f, q));
        qx[base + col] = (_Float16)(q * scale);
    }
}

// ---------------------------------------------------------------------------
// Kernel 2: fake-quantize W per (32,32) block (max-abs is transpose-invariant,
// so native [Dout, Din] layout == blocks of W^T). Output f16, K-contiguous.
// ---------------------------------------------------------------------------
__global__ __launch_bounds__(256)
void quant_w_kernel(const float* __restrict__ w, _Float16* __restrict__ qw, int K) {
    __shared__ float smax[8];
    const int eb = blockIdx.x;
    const int kb = blockIdx.y;
    const int t  = threadIdx.x;
    const int el = t >> 3;
    const int kl = (t & 7) * 4;
    const long idx = (long)(eb * 32 + el) * K + kb * 32 + kl;
    const float4 v = *(const float4*)(w + idx);
    float a = fmaxf(fmaxf(fabsf(v.x), fabsf(v.y)), fmaxf(fabsf(v.z), fabsf(v.w)));
    #pragma unroll
    for (int off = 16; off > 0; off >>= 1)
        a = fmaxf(a, __shfl_xor(a, off, 32));
    if ((t & 31) == 0) smax[t >> 5] = a;
    __syncthreads();
    float m = smax[0];
    #pragma unroll
    for (int i = 1; i < 8; ++i) m = fmaxf(m, smax[i]);
    const float scale = (m > 0.f) ? m * (1.0f / 127.0f) : 1.0f;
    float r[4] = {v.x, v.y, v.z, v.w};
    _Float16 o[4];
    #pragma unroll
    for (int i = 0; i < 4; ++i) {
        float q = rintf(r[i] / scale);
        q = fminf(127.f, fmaxf(-127.f, q));
        o[i] = (_Float16)(q * scale);
    }
    *(ushort4*)(qw + idx) = *(ushort4*)o;
}

// ---------------------------------------------------------------------------
// Async 64B copy global->LDS: 4x global_load_async_to_lds_b128.
// IOFFSET applies to both LDS and global sides (ISA 10.2 async pseudocode),
// so one LDS addr + one 64-bit global addr covers all four chunks.
// Tracked by ASYNCcnt (no VGPR staging -> no register spills).
// ---------------------------------------------------------------------------
__device__ __forceinline__ void async_copy64(unsigned lds, unsigned long long gaddr) {
    asm volatile(
        "global_load_async_to_lds_b128 %0, %1, off\n\t"
        "global_load_async_to_lds_b128 %0, %1, off offset:16\n\t"
        "global_load_async_to_lds_b128 %0, %1, off offset:32\n\t"
        "global_load_async_to_lds_b128 %0, %1, off offset:48"
        :: "v"(lds), "v"(gaddr) : "memory");
}
__device__ __forceinline__ void wait_async_le8() {
    asm volatile("s_wait_asynccnt 0x8" ::: "memory");
}
__device__ __forceinline__ void wait_async_0() {
    asm volatile("s_wait_asynccnt 0x0" ::: "memory");
}

// ---------------------------------------------------------------------------
// Kernel 3: f16 WMMA GEMM  out[M,N] = qx[M,K] * qw[N,K]^T + bias
// 128x128 tile, BK=64, double-buffered LDS filled by async DMA,
// 8 waves (4M x 2N), wave tile 32x64 = 2x4 v_wmma_f32_16x16x32_f16.
// ---------------------------------------------------------------------------
#define BM 128
#define BN 128
#define BK 64
#define LDT 72   // padded LDS row stride in halves (144B), keeps 16B alignment

__global__ __launch_bounds__(256)
void gemm_f16_wmma_kernel(const _Float16* __restrict__ qx,
                          const _Float16* __restrict__ qw,
                          const float* __restrict__ bias,
                          float* __restrict__ out,
                          int M, int N, int K) {
    __shared__ _Float16 sm[2][2][BM * LDT];   // [buf][A/B][tile]

    const int t    = threadIdx.x;
    const int lane = t & 31;
    const int wave = t >> 5;
    const int wm   = wave & 3;     // 4 M slots of 32 rows
    const int wn   = wave >> 2;    // 2 N slots of 64 cols
    const int m0   = blockIdx.y * BM;
    const int n0   = blockIdx.x * BN;

    // ---- tile copy mapping: thread t moves 64B of A and 64B of B per k-step
    const int crow = t >> 1;            // 0..127
    const int coff = (t & 1) * 32;      // half offset within BK
    const _Float16* gA = qx + (long)(m0 + crow) * K + coff;
    const _Float16* gB = qw + (long)(n0 + crow) * K + coff;
    // LDS byte offsets (generic LDS pointer low 32 bits == LDS offset)
    const unsigned ldsA0 = (unsigned)(uintptr_t)&sm[0][0][crow * LDT + coff];
    const unsigned ldsB0 = (unsigned)(uintptr_t)&sm[0][1][crow * LDT + coff];
    const unsigned bufstride = (unsigned)(sizeof(_Float16) * 2 * BM * LDT);

    auto issue_tile = [&](int buf, int k0) {
        async_copy64(ldsA0 + buf * bufstride, (unsigned long long)(gA + k0));
        async_copy64(ldsB0 + buf * bufstride, (unsigned long long)(gB + k0));
    };

    const int fl = lane & 15;   // row/col within 16x16 frag
    const int fh = lane >> 4;   // half-select per ISA 16-bit A/B layout

    v8f acc[2][4];
    const v8f vzero = {};
    #pragma unroll
    for (int i = 0; i < 2; ++i)
        #pragma unroll
        for (int j = 0; j < 4; ++j) acc[i][j] = vzero;

    issue_tile(0, 0);   // prologue: tile 0 -> buf 0

    const int steps = K / BK;
    #pragma unroll 1
    for (int ks = 0; ks < steps; ++ks) {
        const bool has = (ks + 1 < steps);
        if (has) {
            issue_tile((ks + 1) & 1, (ks + 1) * BK);
            if (ks + 2 < steps) {   // warm L2 two tiles ahead
                __builtin_prefetch(gA + (ks + 2) * BK, 0, 1);
                __builtin_prefetch(gB + (ks + 2) * BK, 0, 1);
            }
            wait_async_le8();       // newest 8 outstanding => tile ks landed
        } else {
            wait_async_0();
        }
        __syncthreads();

        const int buf = ks & 1;
        #pragma unroll
        for (int kk = 0; kk < BK; kk += 32) {
            union Frag { v16h v; v8h h[2]; };
            Frag af[2], bf[4];
            #pragma unroll
            for (int i = 0; i < 2; ++i) {
                const _Float16* p = &sm[buf][0][(wm * 32 + i * 16 + fl) * LDT + kk + fh * 8];
                af[i].h[0] = *(const v8h*)p;
                af[i].h[1] = *(const v8h*)(p + 16);
            }
            #pragma unroll
            for (int j = 0; j < 4; ++j) {
                const _Float16* p = &sm[buf][1][(wn * 64 + j * 16 + fl) * LDT + kk + fh * 8];
                bf[j].h[0] = *(const v8h*)p;
                bf[j].h[1] = *(const v8h*)(p + 16);
            }
            #pragma unroll
            for (int i = 0; i < 2; ++i)
                #pragma unroll
                for (int j = 0; j < 4; ++j)
                    acc[i][j] = __builtin_amdgcn_wmma_f32_16x16x32_f16(
                        false, af[i].v, false, bf[j].v,
                        (short)0, acc[i][j], false, false);
        }
        __syncthreads();   // all waves done reading buf before it is re-filled
    }

    // ---- epilogue: lanes 0-15: M=v, N=lane; lanes 16-31: M=v+8
    #pragma unroll
    for (int i = 0; i < 2; ++i) {
        #pragma unroll
        for (int j = 0; j < 4; ++j) {
            const int col  = n0 + wn * 64 + j * 16 + fl;
            const int row0 = m0 + wm * 32 + i * 16 + fh * 8;
            const float bv = bias[col];
            #pragma unroll
            for (int v = 0; v < 8; ++v)
                out[(long)(row0 + v) * N + col] = acc[i][j][v] + bv;
        }
    }
}

// ---------------------------------------------------------------------------
extern "C" void kernel_launch(void* const* d_in, const int* in_sizes, int n_in,
                              void* d_out, int out_size, void* d_ws, size_t ws_size,
                              hipStream_t stream) {
    const float* x    = (const float*)d_in[0];   // [B,S,Din] f32
    const float* w    = (const float*)d_in[1];   // [Dout,Din] f32
    const float* bias = (const float*)d_in[2];   // [Dout] f32
    float* out        = (float*)d_out;           // [B,S,Dout] f32

    const int Dout = in_sizes[2];
    const int Din  = in_sizes[1] / Dout;
    const int M    = in_sizes[0] / Din;

    _Float16* qx = (_Float16*)d_ws;               // M*Din halfs
    _Float16* qw = qx + (size_t)M * Din;          // Dout*Din halfs

    quant_x_kernel<<<M, 256, 0, stream>>>(x, qx, Din);

    dim3 gw(Dout / 32, Din / 32);
    quant_w_kernel<<<gw, 256, 0, stream>>>(w, qw, Din);

    dim3 gg(Dout / BN, M / BM);
    gemm_f16_wmma_kernel<<<gg, 256, 0, stream>>>(qx, qw, bias, out, M, Dout, Din);
}
